// BiasedMHA_55216099558096
// MI455X (gfx1250) — compile-verified, matrix operand order checked by reading, not run on hardware
//
#include <hip/hip_runtime.h>
#include <hip/hip_bf16.h>

typedef __attribute__((ext_vector_type(16))) __bf16 v16bf;
typedef __attribute__((ext_vector_type(8)))  __bf16 v8bf;
typedef __attribute__((ext_vector_type(4)))  __bf16 v4bf;
typedef __attribute__((ext_vector_type(8)))  float  v8f;
typedef __attribute__((ext_vector_type(4)))  unsigned int v4u;
typedef __attribute__((ext_vector_type(8)))  int    v8i;
typedef __attribute__((ext_vector_type(4)))  int    v4i;

#define BB 4
#define NN 1024
#define CC 1024
#define HH 16
#define DD 64

#if defined(__gfx1250__) && __has_builtin(__builtin_amdgcn_tensor_load_to_lds)
#define USE_TDM 1
#else
#define USE_TDM 0
#endif

#if __has_builtin(__builtin_amdgcn_s_wait_tensorcnt)
#define WAIT_TENSORCNT(n) __builtin_amdgcn_s_wait_tensorcnt(n)
#else
#define WAIT_TENSORCNT(n) asm volatile("s_wait_tensorcnt %0" ::"i"(n) : "memory")
#endif

// Build a 16x32 (A) / 32x16 (B) bf16 WMMA fragment for this lane from two
// contiguous 16-byte chunks (ISA 7.12.2: lane holds K = {8h..8h+7} U {16+8h..16+8h+7}).
static __device__ __forceinline__ v16bf ldfrag(const __bf16* lo, const __bf16* hi) {
  v8bf a = *(const v8bf*)lo;
  v8bf b = *(const v8bf*)hi;
  v16bf r;
#pragma unroll
  for (int i = 0; i < 8; ++i) { r[i] = a[i]; r[i + 8] = b[i]; }
  return r;
}

static __device__ __forceinline__ v8f wmma_bf16(v16bf a, v16bf b, v8f c) {
  // D = A(16x32) * B(32x16) + C, f32 accumulate
  return __builtin_amdgcn_wmma_f32_16x16x32_bf16(false, a, false, b, (short)0, c,
                                                 false, false);
}

#if USE_TDM
// 32-bit LDS byte offset of a __shared__ object (generic -> AS(3) -> int).
static __device__ __forceinline__ unsigned lds_addr_of(const void* p) {
  return (unsigned)(unsigned long long)(__attribute__((address_space(3))) const void*)p;
}

// TDM DMA of a 2D bf16 tile (tile_w elems x tile_h rows, row stride in elems)
// from global memory into LDS. pad_enable inserts 4 DWORDs (16 B) after every
// 16 DWORDs (64 B = one 32-elem row) -> matches the stride-40 LDS layout.
// Descriptor per CDNA5 ISA ch.8 (D# groups); all inputs are wave-uniform.
static __device__ __forceinline__ void tdm_load_tile_2d(unsigned lds_byte,
                                                        const void* gtile,
                                                        int tile_w, int tile_h,
                                                        int stride_elems) {
  const unsigned long long ga = (unsigned long long)gtile;
  v4u g0;
  g0[0] = 1u;                                                   // count=1, user desc
  g0[1] = lds_byte;                                             // lds_addr
  g0[2] = (unsigned)ga;                                         // global_addr[31:0]
  g0[3] = ((unsigned)(ga >> 32) & 0x01FFFFFFu) | (2u << 30);    // ga[56:32] | type=2
  v8i g1;
  g1[0] = (1 << 16)                 // data_size = 2 bytes
        | (1 << 20)                 // pad_enable
        | (3 << 22)                 // pad_interval: 16 DWORDs (64 B)
        | (3 << 25);                // pad_amount:   4 DWORDs (16 B)
  g1[1] = (tile_w & 0xFFFF) << 16;  // tensor_dim0 = tile_w (in-bounds tile)
  g1[2] = (tile_h & 0xFFFF) << 16;  // tensor_dim0 hi = 0, tensor_dim1 = tile_h
  g1[3] = (tile_w & 0xFFFF) << 16;  // tensor_dim1 hi = 0, tile_dim0 = tile_w
  g1[4] = (tile_h & 0xFFFF);        // tile_dim1 = tile_h, tile_dim2 = 0
  g1[5] = stride_elems;             // tensor_dim0_stride[31:0]
  g1[6] = 0;                        // stride hi, tensor_dim1_stride lo
  g1[7] = 0;
  v4i g2;
  g2[0] = 0; g2[1] = 0; g2[2] = 0; g2[3] = 0;
  v4i g3 = g2;
#if __clang_major__ >= 23
  v8i g4;
#pragma unroll
  for (int i = 0; i < 8; ++i) g4[i] = 0;
  __builtin_amdgcn_tensor_load_to_lds(g0, g1, g2, g3, g4, 0);
#else
  __builtin_amdgcn_tensor_load_to_lds(g0, g1, g2, g3, 0);
#endif
}
#endif  // USE_TDM

// ---------------------------------------------------------------------------
// fp32 -> bf16 conversion (vectorized x4)
// ---------------------------------------------------------------------------
__global__ __launch_bounds__(256) void cvt_f32_to_bf16(const float* __restrict__ in,
                                                       __bf16* __restrict__ out,
                                                       int n4) {
  int i = blockIdx.x * 256 + threadIdx.x;
  if (i < n4) {
    float4 v = ((const float4*)in)[i];
    v4bf o;
    o[0] = (__bf16)v.x; o[1] = (__bf16)v.y; o[2] = (__bf16)v.z; o[3] = (__bf16)v.w;
    ((v4bf*)out)[i] = o;
  }
}

// ---------------------------------------------------------------------------
// GEMM: out(M=4096, N=1024) = A(M,K=1024) * W(N,K)^T + bias, bf16 in / f32 acc.
// Block tile 128x64, 8 waves (4x2), each wave 32x32 (2x2 WMMA tiles).
// Tiles are staged into LDS by the Tensor Data Mover (double-buffered, one
// issuing wave, TENSORcnt-tracked) when available; sync global->LDS otherwise.
// mode 0: Q  -> (B,H,N,D) bf16, scaled      mode 1: K -> (B,H,N,D) bf16
// mode 2: V  -> (B,H,D,N) bf16 (transposed) mode 3: fp32 row-major (final proj)
// ---------------------------------------------------------------------------
__global__ __launch_bounds__(256) void gemm_wmma(const __bf16* __restrict__ A,
                                                 const __bf16* __restrict__ W,
                                                 const float* __restrict__ bias,
                                                 __bf16* __restrict__ outb,
                                                 float* __restrict__ outf,
                                                 int mode, float scale) {
  __shared__ __bf16 sA[2][128 * 40];  // stride 40 elems (80 B) -> conflict-free
  __shared__ __bf16 sB[2][64 * 40];
  const int tid  = threadIdx.x;
  const int lane = tid & 31, wave = tid >> 5;
  const int hf   = lane >> 4, ln = lane & 15;
  const int waveM = wave & 3, waveN = wave >> 2;
  const int bm = blockIdx.x * 128, bn = blockIdx.y * 64;

  v8f acc[2][2];
#pragma unroll
  for (int i = 0; i < 2; i++)
#pragma unroll
    for (int j = 0; j < 2; j++)
#pragma unroll
      for (int e = 0; e < 8; e++) acc[i][j][e] = 0.0f;

#if USE_TDM
  const bool issuer = (wave == 0);
  const __bf16* gA0 = A + (size_t)bm * CC;
  const __bf16* gB0 = W + (size_t)bn * CC;
  const unsigned ldsA[2] = {lds_addr_of(&sA[0][0]), lds_addr_of(&sA[1][0])};
  const unsigned ldsB[2] = {lds_addr_of(&sB[0][0]), lds_addr_of(&sB[1][0])};
  if (issuer) {                                   // prologue: DMA tile 0
    tdm_load_tile_2d(ldsA[0], gA0, 32, 128, CC);
    tdm_load_tile_2d(ldsB[0], gB0, 32, 64, CC);
  }
#else
  const int arow = tid >> 1, acol = (tid & 1) * 16;  // 128x32 A tile
  const int brow = tid >> 2, bcol = (tid & 3) * 8;   // 64x32  B tile
  const __bf16* ga = A + (size_t)(bm + arow) * CC + acol;
  const __bf16* gb = W + (size_t)(bn + brow) * CC + bcol;
#endif

  for (int it = 0; it < CC / 32; ++it) {
    const int k0 = it * 32;
#if USE_TDM
    const int cur = it & 1;
    if (issuer) {
      if (k0 + 32 < CC) {                         // DMA next tile into other buffer
        tdm_load_tile_2d(ldsA[cur ^ 1], gA0 + k0 + 32, 32, 128, CC);
        tdm_load_tile_2d(ldsB[cur ^ 1], gB0 + k0 + 32, 32, 64, CC);
        WAIT_TENSORCNT(2);                        // current tile's 2 DMAs done
      } else {
        WAIT_TENSORCNT(0);
      }
    }
    __syncthreads();                              // current tile visible to all
    const __bf16* tA = sA[cur];
    const __bf16* tB = sB[cur];
#else
    __syncthreads();
    *(v8bf*)(sA[0] + arow * 40 + acol)     = *(const v8bf*)(ga + k0);
    *(v8bf*)(sA[0] + arow * 40 + acol + 8) = *(const v8bf*)(ga + k0 + 8);
    *(v8bf*)(sB[0] + brow * 40 + bcol)     = *(const v8bf*)(gb + k0);
    if (k0 + 32 < CC) {                           // global_prefetch_b8 next tile
      __builtin_prefetch(ga + k0 + 32, 0, 1);
      __builtin_prefetch(gb + k0 + 32, 0, 1);
    }
    __syncthreads();
    const __bf16* tA = sA[0];
    const __bf16* tB = sB[0];
#endif

    v16bf af[2], bfr[2];
#pragma unroll
    for (int tm = 0; tm < 2; tm++) {
      const __bf16* rp = tA + (waveM * 32 + tm * 16 + ln) * 40;
      af[tm] = ldfrag(rp + 8 * hf, rp + 16 + 8 * hf);
    }
#pragma unroll
    for (int tn = 0; tn < 2; tn++) {
      const __bf16* rp = tB + (waveN * 32 + tn * 16 + ln) * 40;
      bfr[tn] = ldfrag(rp + 8 * hf, rp + 16 + 8 * hf);
    }
#pragma unroll
    for (int tm = 0; tm < 2; tm++)
#pragma unroll
      for (int tn = 0; tn < 2; tn++)
        acc[tm][tn] = wmma_bf16(af[tm], bfr[tn], acc[tm][tn]);

    __syncthreads();                              // done reading before reuse
  }

  // Epilogue: C/D layout row = r + 8*hf, col = ln (ISA 7.12.2)
#pragma unroll
  for (int tm = 0; tm < 2; tm++) {
#pragma unroll
    for (int tn = 0; tn < 2; tn++) {
      const int gn = bn + waveN * 32 + tn * 16 + ln;
      const float bv = bias[gn];
#pragma unroll
      for (int r = 0; r < 8; r++) {
        const int gm = bm + waveM * 32 + tm * 16 + r + 8 * hf;
        const float v = (acc[tm][tn][r] + bv) * scale;
        if (mode == 3) {
          outf[(size_t)gm * CC + gn] = v;
        } else {
          const int bb = gm >> 10, tok = gm & (NN - 1);
          const int hh = gn >> 6,  dd  = gn & (DD - 1);
          size_t idx;
          if (mode == 2) idx = ((size_t)(bb * HH + hh) * DD + dd) * NN + tok;  // V^T
          else           idx = ((size_t)(bb * HH + hh) * NN + tok) * DD + dd;  // Q,K
          outb[idx] = (__bf16)v;
        }
      }
    }
  }
}

// ---------------------------------------------------------------------------
// Fused flash attention: grid (N/128, H, B), 8 waves/block, 16 queries/wave.
// Streams 32 keys/iter; bias read exactly once; online softmax in registers;
// P transposed through per-wave LDS for the PV wmma.
// ---------------------------------------------------------------------------
__global__ __launch_bounds__(256) void flash_wmma(const __bf16* __restrict__ Q,
                                                  const __bf16* __restrict__ Kt,
                                                  const __bf16* __restrict__ Vt,
                                                  const float* __restrict__ bias,
                                                  const int* __restrict__ mask,
                                                  __bf16* __restrict__ Oout) {
  __shared__ __bf16 sP[8][16 * 40];   // per-wave 16x32 P tile, padded stride 40
  const int tid  = threadIdx.x;
  const int lane = tid & 31, wave = tid >> 5;
  const int hf   = lane >> 4, ln = lane & 15;
  const int h = blockIdx.y, b = blockIdx.z;
  const int qbase = blockIdx.x * 128 + wave * 16;
  const size_t bh = (size_t)(b * HH + h);

  // Q fragments (D=64 -> two K=32 chunks), loaded straight from global (b128s)
  const __bf16* qrow = Q + (bh * NN + qbase + ln) * DD;
  const v16bf qa0 = ldfrag(qrow +      8 * hf, qrow + 16 + 8 * hf);
  const v16bf qa1 = ldfrag(qrow + 32 + 8 * hf, qrow + 48 + 8 * hf);

  v8f o[4];
  float mrow[8], lrow[8];
#pragma unroll
  for (int c = 0; c < 4; c++)
#pragma unroll
    for (int e = 0; e < 8; e++) o[c][e] = 0.0f;
#pragma unroll
  for (int r = 0; r < 8; r++) { mrow[r] = -3.0e38f; lrow[r] = 0.0f; }

  for (int j = 0; j < NN; j += 32) {
    // ---- S = Q K^T for 32 keys (two 16-key column tiles) ----
    v8f s[2];
#pragma unroll
    for (int kt = 0; kt < 2; kt++) {
      const __bf16* krow = Kt + (bh * NN + j + kt * 16 + ln) * DD;
      v16bf kb0 = ldfrag(krow +      8 * hf, krow + 16 + 8 * hf);
      v16bf kb1 = ldfrag(krow + 32 + 8 * hf, krow + 48 + 8 * hf);
      v8f z;
#pragma unroll
      for (int e = 0; e < 8; e++) z[e] = 0.0f;
      z = wmma_bf16(qa0, kb0, z);
      z = wmma_bf16(qa1, kb1, z);
      s[kt] = z;
    }

    // ---- + bias, mask ----
    const int mk0 = mask[b * NN + j + ln];
    const int mk1 = mask[b * NN + j + 16 + ln];
#pragma unroll
    for (int r = 0; r < 8; r++) {
      const float* bp = bias + (bh * NN + qbase + r + 8 * hf) * NN + j;
      const float s0 = s[0][r] + bp[ln];
      const float s1 = s[1][r] + bp[16 + ln];
      s[0][r] = mk0 ? s0 : -3.0e38f;
      s[1][r] = mk1 ? s1 : -3.0e38f;
    }

    // ---- online softmax (row reductions over the 16-lane half-group) ----
#pragma unroll
    for (int r = 0; r < 8; r++) {
      float t = fmaxf(s[0][r], s[1][r]);
      t = fmaxf(t, __shfl_xor(t, 1));
      t = fmaxf(t, __shfl_xor(t, 2));
      t = fmaxf(t, __shfl_xor(t, 4));
      t = fmaxf(t, __shfl_xor(t, 8));
      const float mn   = fmaxf(mrow[r], t);
      const float corr = __expf(mrow[r] - mn);
      mrow[r] = mn;
      const float p0 = __expf(s[0][r] - mn);
      const float p1 = __expf(s[1][r] - mn);
      float ps = p0 + p1;
      ps += __shfl_xor(ps, 1);
      ps += __shfl_xor(ps, 2);
      ps += __shfl_xor(ps, 4);
      ps += __shfl_xor(ps, 8);
      lrow[r] = lrow[r] * corr + ps;
#pragma unroll
      for (int c = 0; c < 4; c++) o[c][r] *= corr;
      // write P (D-layout) into LDS for transpose to A-layout
      __bf16* prow = &sP[wave][(r + 8 * hf) * 40];
      prow[ln]      = (__bf16)p0;
      prow[16 + ln] = (__bf16)p1;
    }

    // intra-wave LDS RAW: drain DS counter before re-reading transposed P
    asm volatile("s_wait_dscnt 0x0" ::: "memory");

    const __bf16* pr = &sP[wave][ln * 40];
    const v16bf pa = ldfrag(pr + 8 * hf, pr + 16 + 8 * hf);

    // ---- O += P V  (V stored transposed: rows are dims, contiguous in keys) ----
#pragma unroll
    for (int c = 0; c < 4; c++) {
      const __bf16* vrow = Vt + (bh * DD + c * 16 + ln) * NN + j;
      v16bf vb = ldfrag(vrow + 8 * hf, vrow + 16 + 8 * hf);
      o[c] = wmma_bf16(pa, vb, o[c]);
    }
  }

  // ---- normalize and store bf16 attention output in (B,N,C) layout ----
#pragma unroll
  for (int r = 0; r < 8; r++) {
    const float inv = 1.0f / fmaxf(lrow[r], 1e-30f);
    const size_t orow = ((size_t)b * NN + qbase + r + 8 * hf) * CC + h * DD;
#pragma unroll
    for (int c = 0; c < 4; c++)
      Oout[orow + c * 16 + ln] = (__bf16)(o[c][r] * inv);
  }
}

// ---------------------------------------------------------------------------
extern "C" void kernel_launch(void* const* d_in, const int* in_sizes, int n_in,
                              void* d_out, int out_size, void* d_ws, size_t ws_size,
                              hipStream_t stream) {
  (void)in_sizes; (void)n_in; (void)out_size; (void)ws_size;
  const float* x         = (const float*)d_in[0];
  const float* attn_bias = (const float*)d_in[1];
  const int*   attn_mask = (const int*)d_in[2];
  const float* Wq = (const float*)d_in[3];
  const float* bq = (const float*)d_in[4];
  const float* Wk = (const float*)d_in[5];
  const float* bk = (const float*)d_in[6];
  const float* Wv = (const float*)d_in[7];
  const float* bv = (const float*)d_in[8];
  const float* Wo = (const float*)d_in[9];
  const float* bo = (const float*)d_in[10];

  char* ws = (char*)d_ws;
  const size_t MiB = 1ull << 20;
  __bf16* xb    = (__bf16*)(ws + 0);        // 8 MiB, x in bf16
  __bf16* attnb = (__bf16*)(ws + 0);        // reused after QKV GEMMs complete
  __bf16* wqb   = (__bf16*)(ws + 8  * MiB);
  __bf16* wkb   = (__bf16*)(ws + 10 * MiB);
  __bf16* wvb   = (__bf16*)(ws + 12 * MiB);
  __bf16* wob   = (__bf16*)(ws + 14 * MiB);
  __bf16* Qb    = (__bf16*)(ws + 16 * MiB); // (B,H,N,D) pre-scaled by D^-0.5
  __bf16* Kb    = (__bf16*)(ws + 24 * MiB); // (B,H,N,D)
  __bf16* Vtb   = (__bf16*)(ws + 32 * MiB); // (B,H,D,N)

  cvt_f32_to_bf16<<<4096, 256, 0, stream>>>(x,  xb,  (BB * NN * CC) / 4);
  cvt_f32_to_bf16<<<1024, 256, 0, stream>>>(Wq, wqb, (CC * CC) / 4);
  cvt_f32_to_bf16<<<1024, 256, 0, stream>>>(Wk, wkb, (CC * CC) / 4);
  cvt_f32_to_bf16<<<1024, 256, 0, stream>>>(Wv, wvb, (CC * CC) / 4);
  cvt_f32_to_bf16<<<1024, 256, 0, stream>>>(Wo, wob, (CC * CC) / 4);

  dim3 ggrid(32, 16);  // M=4096 / 128, N=1024 / 64
  gemm_wmma<<<ggrid, 256, 0, stream>>>(xb, wqb, bq, Qb,  nullptr, 0, 0.125f);
  gemm_wmma<<<ggrid, 256, 0, stream>>>(xb, wkb, bk, Kb,  nullptr, 1, 1.0f);
  gemm_wmma<<<ggrid, 256, 0, stream>>>(xb, wvb, bv, Vtb, nullptr, 2, 1.0f);

  flash_wmma<<<dim3(NN / 128, HH, BB), 256, 0, stream>>>(Qb, Kb, Vtb, attn_bias,
                                                         attn_mask, attnb);

  gemm_wmma<<<ggrid, 256, 0, stream>>>(attnb, wob, bo, nullptr, (float*)d_out,
                                       3, 1.0f);
}
